// ShortPosEncodedDecoder_7318624272847
// MI455X (gfx1250) — compile-verified, matrix operand order checked by reading, not run on hardware
//
#include <hip/hip_runtime.h>
#include <cstddef>
#include <cstdint>

// ============================================================================
// ShortPosEncodedDecoder for MI455X (gfx1250, wave32, WMMA, TDM).
//
// Roofline: 70MB fp32 weights @ 23.3TB/s => ~3us floor; M=128 rows gives
// AI = 64 FLOP/B << machine balance => GEMMs are HBM-bound on weights.
// => full-precision V_WMMA_F32_16X16X4_F32 (no accuracy loss, no speed loss),
//    weights streamed into LDS by the Tensor Data Mover (double-buffered,
//    TENSORcnt-synchronized), hardware row padding -> bank-conflict-free.
// Oscillator = 67M sin + wave32 scans fused in registers.
// Noise branch = circular convolution (irfft(rfft(x)*H) == x (*) irfft(H)).
// ============================================================================

typedef float v2f __attribute__((ext_vector_type(2)));
typedef float v8f __attribute__((ext_vector_type(8)));
#if defined(__gfx1250__)
typedef __bf16 v16bf __attribute__((ext_vector_type(16)));
#endif
typedef unsigned int u32x4 __attribute__((ext_vector_type(4)));
typedef int i32x8 __attribute__((ext_vector_type(8)));
typedef int i32x4 __attribute__((ext_vector_type(4)));

#ifndef __has_builtin
#define __has_builtin(x) 0
#endif

#if defined(__gfx1250__) && __has_builtin(__builtin_amdgcn_tensor_load_to_lds) && \
    __has_builtin(__builtin_amdgcn_s_wait_tensorcnt)
#define HAVE_TDM 1
#else
#define HAVE_TDM 0
#endif

#define BATCH   4
#define C       1024
#define BAND    16384
#define HOP     512
#define WINDOW  1024
#define NCOEF   513
#define NWHITE  16896                 // TOTAL + HOP

// ---------------------------------------------------------------------------
// Gather A_lat[m][c] = x[b][c][t/4]   (m = b*32 + t)
// ---------------------------------------------------------------------------
__global__ __launch_bounds__(256) void gather_lat_kernel(
    const float* __restrict__ x, float* __restrict__ A_lat) {
  int idx = blockIdx.x * 256 + threadIdx.x;      // 0 .. 131071
  int m = idx >> 10, c = idx & 1023;
  int b = m >> 5, t = m & 31;
  A_lat[(size_t)m * C + c] = x[(size_t)b * C * 8 + (size_t)c * 8 + (t >> 2)];
}

// ---------------------------------------------------------------------------
// Positional-feature mini-GEMM (K=33): h0[m][1024+o] = feats(t) . pos_w[o] + b
// ---------------------------------------------------------------------------
__global__ __launch_bounds__(256) void pos_kernel(
    const float* __restrict__ pos_w, const float* __restrict__ pos_b,
    float* __restrict__ h0) {
  int idx = blockIdx.x * 256 + threadIdx.x;      // 0 .. 131071
  int m = idx >> 10, o = idx & 1023;
  int t = m & 31;
  float z = -1.0f + 2.0f * (float)t / 31.0f;
  const float* w = pos_w + (size_t)o * 33;
  float acc = w[0] * z;
  float p = 1.0f;
  #pragma unroll
  for (int i = 0; i < 16; ++i) {
    float a = p * z;
    acc += w[1 + 2 * i] * __sinf(a) + w[2 + 2 * i] * __cosf(a);
    p *= 2.0f;
  }
  h0[(size_t)m * 2048 + 1024 + o] = acc + pos_b[o];
}

// ---------------------------------------------------------------------------
// TDM descriptor helpers (D# groups per CDNA5 ISA).
// 2D tile of 4-byte elements, LDS row padding: +4 dwords after every 32
// dwords stored => LDS row stride 36 floats (bank-conflict-free, and matches
// the fragment access pattern of the f32 16x16x4 WMMA).
// NOTE: this toolchain's builtin is the 6-arg (clang-23 / therock) form:
//   (uint32x4 g0, int32x8 g1, int32x4 g2, int32x4 g3, int32x8 ext, i32 cpol)
// ---------------------------------------------------------------------------
#if HAVE_TDM
__device__ __forceinline__ unsigned lds_offset_of(const void* p) {
  // Generic LDS address = {SHARED_BASE aperture, 32-bit LDS offset}:
  // low 32 bits are the wave-relative LDS byte offset.
  return (unsigned)(uintptr_t)p;
}

__device__ __forceinline__ void tdm_load_tile_f32(
    const float* gptr, unsigned lds_off,
    unsigned tensor_d0, unsigned tensor_d1,
    unsigned tile_d0, unsigned tile_d1, unsigned stride0_elems) {
  unsigned long long ga = (unsigned long long)(uintptr_t)gptr;
  u32x4 g0;
  g0[0] = 1u;                                       // count=1, user mode
  g0[1] = lds_off;                                  // lds_addr (bytes)
  g0[2] = (unsigned)(ga & 0xFFFFFFFFu);             // global_addr[31:0]
  g0[3] = (unsigned)((ga >> 32) & 0x1FFFFFFu)       // global_addr[56:32]
          | (2u << 30);                             // type=2 ("image")
  i32x8 g1;
  g1[0] = (int)((2u << 16)                          // data_size = 4B
                | (1u << 20)                        // pad_enable
                | (4u << 22)                        // pad_interval: 32 dwords
                | (3u << 25));                      // pad_amount: 4 dwords
  g1[1] = (int)((tensor_d0 & 0xFFFFu) << 16);       // td0[15:0]
  g1[2] = (int)(((tensor_d0 >> 16) & 0xFFFFu) | ((tensor_d1 & 0xFFFFu) << 16));
  g1[3] = (int)(((tensor_d1 >> 16) & 0xFFFFu) | ((tile_d0 & 0xFFFFu) << 16));
  g1[4] = (int)(tile_d1 & 0xFFFFu);                 // tile_dim2 = 0
  g1[5] = (int)stride0_elems;                       // dim0 stride (elems)
  g1[6] = 0;                                        // stride hi, dim1_stride=0
  g1[7] = 0;
  const i32x4 gz4 = {0, 0, 0, 0};                   // <=2D: groups 2/3 unused
  const i32x8 gz8 = {0, 0, 0, 0, 0, 0, 0, 0};
  __builtin_amdgcn_tensor_load_to_lds(g0, g1, gz4, gz4, gz8, 0);
}
#endif

// ---------------------------------------------------------------------------
// WMMA GEMM: Y(128 x N) = act( X(128 x K) @ W(N x K)^T + bias ).
// Block = 256 threads = 8 waves; each wave owns one 16x16 M-tile of a 16-wide
// N strip.  A/B chunks live in LDS at row stride 36 floats.  TDM path: wave 0
// issues 2 DMA descriptors per chunk into the ping-pong buffer, waits on
// TENSORcnt, barriers; all waves run only WMMA.  Dual accumulators break the
// D->C dependence chain between consecutive WMMAs.
// ---------------------------------------------------------------------------
__global__ __launch_bounds__(256) void gemm128_kernel(
    const float* __restrict__ X, const float* __restrict__ W,
    const float* __restrict__ bias, float* __restrict__ Y,
    int K, int N, int ldy, int lrelu) {
  __shared__ __align__(16) float As[2][128 * 36];
  __shared__ __align__(16) float Bs[2][16 * 36];

  const int tid  = threadIdx.x;
  const int n0   = blockIdx.x * 16;
  const int lane = tid & 31;
  const int lh   = lane >> 4;          // half-wave select
  const int lrow = lane & 15;
  const int m0   = (tid >> 5) * 16;    // wave's M tile

  v8f acc0 = {0.f, 0.f, 0.f, 0.f, 0.f, 0.f, 0.f, 0.f};
  v8f acc1 = acc0;

  const int nch = K >> 5;              // 32-wide K chunks

#if HAVE_TDM
  const bool issuer = (tid < 32);      // wave 0 drives the DMA
  const unsigned bn = (unsigned)((N - n0) < 16 ? (N - n0) : 16);
  if (issuer) {
    tdm_load_tile_f32(X, lds_offset_of(&As[0][0]), 32, 128, 32, 128, (unsigned)K);
    tdm_load_tile_f32(W + (size_t)n0 * K, lds_offset_of(&Bs[0][0]),
                      32, bn, 32, 16, (unsigned)K);
  }
#else
  const int lr = tid >> 3;             // 0..31  (loader row group)
  const int lc = (tid & 7) << 2;       // 0..28  (loader col*4)
#endif

  for (int ic = 0; ic < nch; ++ic) {
    const int buf = ic & 1;
    const float* Ab = As[buf];
    const float* Bb = Bs[buf];

#if HAVE_TDM
    if (issuer) {
      if (ic + 1 < nch) {              // prefetch next chunk into other buffer
        const int kk2 = (ic + 1) << 5;
        tdm_load_tile_f32(X + kk2, lds_offset_of(&As[buf ^ 1][0]),
                          32, 128, 32, 128, (unsigned)K);
        tdm_load_tile_f32(W + (size_t)n0 * K + kk2, lds_offset_of(&Bs[buf ^ 1][0]),
                          32, bn, 32, 16, (unsigned)K);
        __builtin_amdgcn_s_wait_tensorcnt((short)2);  // chunk ic landed
      } else {
        __builtin_amdgcn_s_wait_tensorcnt((short)0);
      }
    }
    __syncthreads();                   // chunk ic visible to all waves
#else
    {
      const int kk = ic << 5;
      #pragma unroll
      for (int rr = 0; rr < 4; ++rr) {
        const int row = lr + (rr << 5);
        const float4 av = *(const float4*)(X + (size_t)row * K + kk + lc);
        *(float4*)(&As[0][row * 36 + lc]) = av;
      }
      if (tid < 128) {
        const int gn = n0 + (tid >> 3);
        float4 bv = make_float4(0.f, 0.f, 0.f, 0.f);
        if (gn < N) {
          bv = *(const float4*)(W + (size_t)gn * K + kk + lc);
          if (kk + 32 < K)             // weight-stream prefetch
            __builtin_prefetch(W + (size_t)gn * K + kk + 32 + lc, 0, 0);
        }
        *(float4*)(&Bs[0][(tid >> 3) * 36 + lc]) = bv;
      }
      Ab = As[0];
      Bb = Bs[0];
    }
    __syncthreads();
#endif

#if defined(__gfx1250__) && __has_builtin(__builtin_amdgcn_wmma_f32_16x16x4_f32)
    // Full-precision f32 matrix pipe: 8 x V_WMMA_F32_16X16X4_F32 per chunk,
    // interleaved over two accumulators to break the D->C RAW chain.
    #pragma unroll
    for (int kq = 0; kq < 32; kq += 8) {
      v2f a0, b0, a1, b1;
      const int ar = (m0 + lrow) * 36 + kq + 2 * lh;
      const int br = lrow * 36 + kq + 2 * lh;
      a0[0] = Ab[ar];     a0[1] = Ab[ar + 1];
      b0[0] = Bb[br];     b0[1] = Bb[br + 1];
      a1[0] = Ab[ar + 4]; a1[1] = Ab[ar + 5];
      b1[0] = Bb[br + 4]; b1[1] = Bb[br + 5];
      acc0 = __builtin_amdgcn_wmma_f32_16x16x4_f32(
          false, a0, false, b0, (short)0, acc0, false, false);
      acc1 = __builtin_amdgcn_wmma_f32_16x16x4_f32(
          false, a1, false, b1, (short)0, acc1, false, false);
    }
#elif defined(__gfx1250__)
    // Fallback: codegen-confirmed bf16 WMMA (16x16x32), on-the-fly convert.
    {
      v16bf a, b;
      #pragma unroll
      for (int v = 0; v < 8; ++v) {
        #pragma unroll
        for (int sl = 0; sl < 2; ++sl) {
          const int kq = (v < 4 ? 0 : 16) + lh * 8 + 2 * (v & 3) + sl;
          a[2 * v + sl] = (__bf16)Ab[(m0 + lrow) * 36 + kq];
          b[2 * v + sl] = (__bf16)Bb[lrow * 36 + kq];
        }
      }
      acc0 = __builtin_amdgcn_wmma_f32_16x16x32_bf16(
          false, a, false, b, (short)0, acc0, false, false);
    }
#else
    // Host pass / other arch: never executed, just keep the parser happy.
    acc0[0] += Ab[m0 * 36] * Bb[lrow * 36];
#endif
    __syncthreads();                   // all waves done with buf before reuse
  }

  // Epilogue: C/D layout -> row = m0 + r + 8*half, col = n0 + lrow.
  #pragma unroll
  for (int r = 0; r < 8; ++r) {
    const int row = m0 + r + 8 * lh;
    const int col = n0 + lrow;
    if (col < N) {
      float v = acc0[r] + acc1[r] + bias[col];
      if (lrelu && v < 0.f) v *= 0.2f;
      Y[(size_t)row * ldy + col] = v;
    }
  }
}

// ---------------------------------------------------------------------------
// Chunk-boundary phase table.  Cumsum of the piecewise-linear upsample has a
// closed form per 512-sample chunk: S_g = 64*F[g-1] + 384*F[g] + 64*F[g+1]
// (indices clamped to [0,31]; identity also covers both edge chunks).
// PH[(b*1024+c)*32 + g] = sum of freq_up over t < 512*g.
// ---------------------------------------------------------------------------
__global__ __launch_bounds__(256) void phase_table_kernel(
    const float* __restrict__ freqC,   // (128 rows x 1024): [b*32+t][c]
    float* __restrict__ PH) {
  int idx = blockIdx.x * 256 + threadIdx.x;   // 0..4095
  int b = idx >> 10, c = idx & 1023;
  float acc = 0.f;
  float Fg   = freqC[(size_t)(b * 32 + 0) * C + c];
  float Fgm1 = Fg;
  float* out = PH + ((size_t)(b << 10) + c) * 32;
  for (int g = 0; g < 32; ++g) {
    int gp1 = (g < 31) ? g + 1 : 31;
    float Fgp1 = freqC[(size_t)(b * 32 + gp1) * C + c];
    out[g] = acc;
    acc += 64.0f * Fgm1 + 384.0f * Fg + 64.0f * Fgp1;
    Fgm1 = Fg;
    Fg = Fgp1;
  }
}

// ---------------------------------------------------------------------------
// Oscillator: out[b][t] = (1/1024) * sum_c sin(phase[b][c][t]) * amp_up[b][c][t]
// Block = (chunk g, batch b): 8 waves x 128 channels each, 512 samples.
// Closed-form interp weights, wave32 inclusive scan via __shfl_up, v_sin_f32,
// LDS ds_add_f32 channel reduction.  Exclusive ownership => plain stores.
// ---------------------------------------------------------------------------
__global__ __launch_bounds__(256) void osc_kernel(
    const float* __restrict__ freqC, const float* __restrict__ ampC,
    const float* __restrict__ PH, float* __restrict__ out) {
  __shared__ float sacc[512];
  const int g = blockIdx.x;            // 0..31
  const int b = blockIdx.y;            // 0..3
  const int tid = threadIdx.x;
  const int wave = tid >> 5, lane = tid & 31;

  for (int i = tid; i < 512; i += 256) sacc[i] = 0.f;
  __syncthreads();

  const int gm1 = (g > 0) ? g - 1 : 0;
  const int gp1 = (g < 31) ? g + 1 : 31;

  float acc[16];
  #pragma unroll
  for (int sb = 0; sb < 16; ++sb) acc[sb] = 0.f;

  for (int ci = 0; ci < 128; ++ci) {
    const int ch = wave + (ci << 3);   // interleave channels across waves
    const size_t r0 = (size_t)(b * 32 + gm1) * C + ch;
    const size_t r1 = (size_t)(b * 32 + g)   * C + ch;
    const size_t r2 = (size_t)(b * 32 + gp1) * C + ch;
    const float f0 = freqC[r0], f1 = freqC[r1], f2 = freqC[r2];
    const float a0 = ampC[r0],  a1 = ampC[r1],  a2 = ampC[r2];
    float carry = PH[((size_t)(b << 10) + ch) * 32 + g];

    #pragma unroll
    for (int sb = 0; sb < 16; ++sb) {
      const int s = (sb << 5) + lane;  // 0..511 within chunk
      float w, flo, fhi, alo, ahi;
      if (sb < 8) {                    // first half: interp F[g-1] -> F[g]
        w = (g == 0) ? 1.0f : 0.5f + ((float)s + 0.5f) * (1.0f / 512.0f);
        flo = f0; fhi = f1; alo = a0; ahi = a1;
      } else {                         // second half: interp F[g] -> F[g+1]
        w = (g == 31) ? 0.0f : ((float)(s - 256) + 0.5f) * (1.0f / 512.0f);
        flo = f1; fhi = f2; alo = a1; ahi = a2;
      }
      float fv = flo + (fhi - flo) * w;
      // wave32 inclusive scan
      float scan = fv;
      #pragma unroll
      for (int d = 1; d < 32; d <<= 1) {
        float o = __shfl_up(scan, d, 32);
        if (lane >= d) scan += o;
      }
      float phase = carry + scan;
      carry = __shfl(carry + scan, 31, 32);   // += block total (broadcast)
      float av = alo + (ahi - alo) * w;
      acc[sb] += __sinf(phase) * av;
    }
  }

  #pragma unroll
  for (int sb = 0; sb < 16; ++sb)
    atomicAdd(&sacc[(sb << 5) + lane], acc[sb]);   // ds_add_f32
  __syncthreads();

  for (int i = tid; i < 512; i += 256)
    out[(size_t)b * BAND + g * HOP + i] = sacc[i] * (1.0f / 1024.0f);
}

// ---------------------------------------------------------------------------
// Noise step 1: h = irfft(H) per frame.
// h[d] = (H[0] + H[512]*cos(pi d) + 2*sum_{k=1}^{511} H[k] cos(2 pi k d/1024))/1024
// ---------------------------------------------------------------------------
__global__ __launch_bounds__(256) void noise_ir_kernel(
    const float* __restrict__ noiseF,  // (128 x 513)
    float* __restrict__ hconv) {       // (128 x 1024)
  __shared__ float Hs[NCOEF];
  const int m = blockIdx.x;            // b*32 + fr
  const int tid = threadIdx.x;
  for (int i = tid; i < NCOEF; i += 256) Hs[i] = noiseF[(size_t)m * NCOEF + i];
  __syncthreads();

  const float c0 = Hs[0], cN = Hs[512];
  #pragma unroll
  for (int j = 0; j < 4; ++j) {
    const int d = tid + (j << 8);
    float sum = c0 + ((d & 1) ? -cN : cN);
    const float step = 6.28318530717958647692f / 1024.0f;
    for (int k = 1; k < 512; ++k)
      sum += 2.0f * Hs[k] * __cosf((float)(k * d) * step);
    hconv[(size_t)m * WINDOW + d] = sum * (1.0f / 1024.0f);
  }
}

// ---------------------------------------------------------------------------
// Noise step 2: circular conv + overlap-add:
//   audio[n] = sum_m frame[m] * h[(n-m) mod 1024]
//   out[b, fr*512 + n]         += nf * audio[n]   (n < 512)
//   out[b, (fr+1)*512 + n-512] += nf * audio[n]   (n >= 512, fr < 31)
// ---------------------------------------------------------------------------
__global__ __launch_bounds__(256) void noise_conv_kernel(
    const float* __restrict__ noise_white, const float* __restrict__ hconv,
    const float* __restrict__ noise_factor, float* __restrict__ out) {
  __shared__ float xs[WINDOW];
  __shared__ float hs[WINDOW];
  const int m = blockIdx.x;            // b*32 + fr
  const int b = m >> 5, fr = m & 31;
  const int tid = threadIdx.x;
  for (int i = tid; i < WINDOW; i += 256) {
    xs[i] = noise_white[(size_t)b * NWHITE + fr * HOP + i];
    hs[i] = hconv[(size_t)m * WINDOW + i];
  }
  __syncthreads();
  const float nf = noise_factor[0];
  #pragma unroll
  for (int j = 0; j < 4; ++j) {
    const int n = tid + (j << 8);
    float sum = 0.f;
    for (int mm = 0; mm < WINDOW; ++mm)
      sum += xs[mm] * hs[(n - mm) & (WINDOW - 1)];
    const float val = sum * nf;
    if (n < HOP) {
      atomicAdd(out + (size_t)b * BAND + fr * HOP + n, val);
    } else if (fr < 31) {
      atomicAdd(out + (size_t)b * BAND + (fr + 1) * HOP + (n - HOP), val);
    }
  }
}

// ===========================================================================
// Launch.  Input order assumes dict-insertion-order flattening of
// setup_inputs(): x, pos_w, pos_b, lat_w, lat_b, tr_ws[0..5], tr_bs[0..5],
// amp_ws[0..2], amp_bs[0..2], freq_ws[0..2], freq_bs[0..2], noise_ws[0..3],
// noise_bs[0..3], noise_factor, noise_white.
// ===========================================================================
extern "C" void kernel_launch(void* const* d_in, const int* in_sizes, int n_in,
                              void* d_out, int out_size, void* d_ws, size_t ws_size,
                              hipStream_t stream) {
  const float* x       = (const float*)d_in[0];
  const float* pos_w   = (const float*)d_in[1];
  const float* pos_b   = (const float*)d_in[2];
  const float* lat_w   = (const float*)d_in[3];
  const float* lat_b   = (const float*)d_in[4];
  const float* tr_w[6];  for (int i = 0; i < 6; ++i) tr_w[i]  = (const float*)d_in[5 + i];
  const float* tr_b[6];  for (int i = 0; i < 6; ++i) tr_b[i]  = (const float*)d_in[11 + i];
  const float* amp_w[3]; for (int i = 0; i < 3; ++i) amp_w[i] = (const float*)d_in[17 + i];
  const float* amp_b[3]; for (int i = 0; i < 3; ++i) amp_b[i] = (const float*)d_in[20 + i];
  const float* frq_w[3]; for (int i = 0; i < 3; ++i) frq_w[i] = (const float*)d_in[23 + i];
  const float* frq_b[3]; for (int i = 0; i < 3; ++i) frq_b[i] = (const float*)d_in[26 + i];
  const float* noi_w[4]; for (int i = 0; i < 4; ++i) noi_w[i] = (const float*)d_in[29 + i];
  const float* noi_b[4]; for (int i = 0; i < 4; ++i) noi_b[i] = (const float*)d_in[33 + i];
  const float* noise_factor = (const float*)d_in[37];
  const float* noise_white  = (const float*)d_in[38];
  float* out = (float*)d_out;

  // Workspace layout (floats)
  float* ws   = (float*)d_ws;
  float* ALAT = ws;                    // 128*1024
  float* H0   = ALAT + 131072;         // 128*2048
  float* B1   = H0   + 262144;         // 128*1024
  float* B2   = B1   + 131072;         // 128*1024
  float* HF   = B2   + 131072;         // 128*1024
  float* AMP  = HF   + 131072;         // 128*1024
  float* FRQ  = AMP  + 131072;         // 128*1024
  float* NF   = FRQ  + 131072;         // 128*513
  float* PH   = NF   + 65664;          // 4*1024*32
  float* HC   = PH   + 131072;         // 128*1024
  (void)in_sizes; (void)n_in; (void)out_size; (void)ws_size;

  const dim3 blk(256);

  // ---- input projections ----
  gather_lat_kernel<<<512, blk, 0, stream>>>(x, ALAT);
  gemm128_kernel<<<64, blk, 0, stream>>>(ALAT, lat_w, lat_b, H0, 1024, 1024, 2048, 0);
  pos_kernel<<<512, blk, 0, stream>>>(pos_w, pos_b, H0);

  // ---- trunk (leaky-relu on all but last) ----
  gemm128_kernel<<<64, blk, 0, stream>>>(H0, tr_w[0], tr_b[0], B1, 2048, 1024, 1024, 1);
  gemm128_kernel<<<64, blk, 0, stream>>>(B1, tr_w[1], tr_b[1], B2, 1024, 1024, 1024, 1);
  gemm128_kernel<<<64, blk, 0, stream>>>(B2, tr_w[2], tr_b[2], B1, 1024, 1024, 1024, 1);
  gemm128_kernel<<<64, blk, 0, stream>>>(B1, tr_w[3], tr_b[3], B2, 1024, 1024, 1024, 1);
  gemm128_kernel<<<64, blk, 0, stream>>>(B2, tr_w[4], tr_b[4], B1, 1024, 1024, 1024, 1);
  gemm128_kernel<<<64, blk, 0, stream>>>(B1, tr_w[5], tr_b[5], HF, 1024, 1024, 1024, 0);

  // ---- heads ----
  gemm128_kernel<<<64, blk, 0, stream>>>(HF, amp_w[0], amp_b[0], B1, 1024, 1024, 1024, 1);
  gemm128_kernel<<<64, blk, 0, stream>>>(B1, amp_w[1], amp_b[1], B2, 1024, 1024, 1024, 1);
  gemm128_kernel<<<64, blk, 0, stream>>>(B2, amp_w[2], amp_b[2], AMP, 1024, 1024, 1024, 0);

  gemm128_kernel<<<64, blk, 0, stream>>>(HF, frq_w[0], frq_b[0], B1, 1024, 1024, 1024, 1);
  gemm128_kernel<<<64, blk, 0, stream>>>(B1, frq_w[1], frq_b[1], B2, 1024, 1024, 1024, 1);
  gemm128_kernel<<<64, blk, 0, stream>>>(B2, frq_w[2], frq_b[2], FRQ, 1024, 1024, 1024, 0);

  gemm128_kernel<<<64, blk, 0, stream>>>(HF, noi_w[0], noi_b[0], B1, 1024, 1024, 1024, 1);
  gemm128_kernel<<<64, blk, 0, stream>>>(B1, noi_w[1], noi_b[1], B2, 1024, 1024, 1024, 1);
  gemm128_kernel<<<64, blk, 0, stream>>>(B2, noi_w[2], noi_b[2], B1, 1024, 1024, 1024, 1);
  gemm128_kernel<<<33, blk, 0, stream>>>(B1, noi_w[3], noi_b[3], NF, 1024, 513, 513, 0);

  // ---- oscillator (owns every out element: plain stores) ----
  phase_table_kernel<<<16, blk, 0, stream>>>(FRQ, PH);
  osc_kernel<<<dim3(32, 4), blk, 0, stream>>>(FRQ, AMP, PH, out);

  // ---- filtered noise, overlap-added on top (atomic adds, after osc) ----
  noise_ir_kernel<<<128, blk, 0, stream>>>(NF, HC);
  noise_conv_kernel<<<128, blk, 0, stream>>>(noise_white, HC, noise_factor, out);
}
